// HeteroGCN_54425825575251
// MI455X (gfx1250) — matrix operand order repaired and checked by reading, not compiled, not run on hardware
//
#include <hip/hip_runtime.h>
#include <hip/hip_bf16.h>

typedef __bf16 bf16;
typedef __attribute__((ext_vector_type(16))) __bf16 v16bf;
typedef __attribute__((ext_vector_type(8)))  __bf16 v8bf;
typedef __attribute__((ext_vector_type(4)))  __bf16 v4bf;
typedef __attribute__((ext_vector_type(8)))  float  v8f;

#define NNODES 100000
#define NEDGES 1600000
#define NREL   3
#define DIN    128
#define DHID   128
#define DOUT   64

// Native f32 atomic add (global_atomic_add_f32). Accumulators are coarse-grained
// device memory and degree counts are exact small integers, so "unsafe" FP atomic
// semantics are fine here.
__device__ __forceinline__ void fadd(float* p, float v) { unsafeAtomicAdd(p, v); }

// ---------------------------------------------------------------------------
// Edge scatter, D=128: one wave32 per edge. Wave-uniform edge metadata is
// forced through readfirstlane so src/dst/dinv become scalar (s_load) traffic;
// the vector pipe does one float4 gather + 4 native f32 atomics per lane.
// ---------------------------------------------------------------------------
__global__ void k_edge_scatter128(const float* __restrict__ xl, const float* __restrict__ dinv,
                                  const int* __restrict__ src, const int* __restrict__ dst,
                                  float* __restrict__ acc, int nedges)
{
    int gw   = (blockIdx.x * blockDim.x + threadIdx.x) >> 5;
    int lane = threadIdx.x & 31;
    if (gw >= nedges) return;
    gw = __builtin_amdgcn_readfirstlane(gw);
    int s = __builtin_amdgcn_readfirstlane(src[gw]);
    int d = __builtin_amdgcn_readfirstlane(dst[gw]);
    float nrm = dinv[s] * dinv[d];
    float4 v = *(const float4*)(xl + (size_t)s * 128 + lane * 4);
    float* ad = acc + (size_t)d * 128 + lane * 4;
    fadd(ad + 0, v.x * nrm);
    fadd(ad + 1, v.y * nrm);
    fadd(ad + 2, v.z * nrm);
    fadd(ad + 3, v.w * nrm);
}

// Edge scatter, D=64: float2 per lane.
__global__ void k_edge_scatter64(const float* __restrict__ xl, const float* __restrict__ dinv,
                                 const int* __restrict__ src, const int* __restrict__ dst,
                                 float* __restrict__ acc, int nedges)
{
    int gw   = (blockIdx.x * blockDim.x + threadIdx.x) >> 5;
    int lane = threadIdx.x & 31;
    if (gw >= nedges) return;
    gw = __builtin_amdgcn_readfirstlane(gw);
    int s = __builtin_amdgcn_readfirstlane(src[gw]);
    int d = __builtin_amdgcn_readfirstlane(dst[gw]);
    float nrm = dinv[s] * dinv[d];
    float2 v = *(const float2*)(xl + (size_t)s * 64 + lane * 2);
    float* ad = acc + (size_t)d * 64 + lane * 2;
    fadd(ad + 0, v.x * nrm);
    fadd(ad + 1, v.y * nrm);
}

// ---------------------------------------------------------------------------
// WMMA GEMM with fused GCN epilogue:
//   xl[n,M]  = A @ B            (bf16 hi/lo split: Ah@Bh + Ah@Be + Ae@Bh)
//   acc[n,M] += dinv[n]^2 * xl[n,M] + bias[col]     (self-loop + bias, fused)
// A: [n,K] bf16 row-major (hi/lo). B: [M,K] bf16 (weights pre-transposed).
// blockDim.x = (M/16)*32, gridDim.x = n/16. Zero divergence: EXEC all-ones.
// ---------------------------------------------------------------------------
__global__ void __launch_bounds__(256)
gcn_gemm_wmma_bf16(const bf16* __restrict__ Ah, const bf16* __restrict__ Ae,
                   const bf16* __restrict__ Bh, const bf16* __restrict__ Be,
                   const float* __restrict__ dinv, const float* __restrict__ bias,
                   float* __restrict__ xl, float* __restrict__ accp, int K, int M)
{
    const int lane = threadIdx.x & 31;
    const int wave = threadIdx.x >> 5;      // column tile
    const int half = lane >> 4;             // 0: lanes 0-15, 1: lanes 16-31
    const int l15  = lane & 15;
    const int row  = blockIdx.x * 16 + l15; // A row this lane feeds
    const int col  = wave * 16 + l15;       // output column this lane owns

    const bf16* a_h = Ah + (size_t)row * K + half * 16;
    const bf16* a_e = Ae + (size_t)row * K + half * 16;
    const bf16* b_h = Bh + (size_t)col * K + half * 16;
    const bf16* b_e = Be + (size_t)col * K + half * 16;

    v8f acc = {};
    for (int k0 = 0; k0 < K; k0 += 32) {
        v8bf ah0 = *(const v8bf*)(a_h + k0);
        v8bf ah1 = *(const v8bf*)(a_h + k0 + 8);
        v8bf ae0 = *(const v8bf*)(a_e + k0);
        v8bf ae1 = *(const v8bf*)(a_e + k0 + 8);
        v16bf A_hi = __builtin_shufflevector(ah0, ah1, 0,1,2,3,4,5,6,7,8,9,10,11,12,13,14,15);
        v16bf A_lo = __builtin_shufflevector(ae0, ae1, 0,1,2,3,4,5,6,7,8,9,10,11,12,13,14,15);
        v16bf B_hi = *(const v16bf*)(b_h + k0);
        v16bf B_lo = *(const v16bf*)(b_e + k0);
        acc = __builtin_amdgcn_wmma_f32_16x16x32_bf16(false, A_hi, false, B_hi, (short)0, acc, false, false);
        acc = __builtin_amdgcn_wmma_f32_16x16x32_bf16(false, A_hi, false, B_lo, (short)0, acc, false, false);
        acc = __builtin_amdgcn_wmma_f32_16x16x32_bf16(false, A_lo, false, B_hi, (short)0, acc, false, false);
    }

    // Fused epilogue: this lane owns rows rowbase..rowbase+7 at column `col`.
    const int rowbase = blockIdx.x * 16 + half * 8;
    const float bcol = bias[col];
    float* c = xl   + (size_t)rowbase * M + col;
    float* a = accp + (size_t)rowbase * M + col;
#pragma unroll
    for (int i = 0; i < 8; ++i) {
        float di = dinv[rowbase + i];
        float v  = acc[i];
        c[(size_t)i * M] = v;
        a[(size_t)i * M] += di * di * v + bcol;
    }
}

// ---------------------------------------------------------------------------
// Degree / norm kernels
// ---------------------------------------------------------------------------
__global__ void k_fill1(float* __restrict__ p, int n) {
    int i = blockIdx.x * blockDim.x + threadIdx.x;
    if (i < n) p[i] = 1.0f;            // self-loop contributes 1 to every degree
}

__global__ void k_deg_count(const int* __restrict__ ei, float* __restrict__ deg,
                            int nedges, int nnodes) {
    int e = blockIdx.x * blockDim.x + threadIdx.x;
    if (e >= nedges) return;
    int r = blockIdx.y;
    int d = ei[(size_t)r * 2 * nedges + nedges + e];     // dst row of relation r
    fadd(&deg[(size_t)r * nnodes + d], 1.0f);
}

__global__ void k_rsqrt(float* __restrict__ p, int n) {
    int i = blockIdx.x * blockDim.x + threadIdx.x;
    if (i < n) { float v = p[i]; p[i] = v > 0.0f ? rsqrtf(v) : 0.0f; }
}

// ---------------------------------------------------------------------------
// Vectorized bf16 hi/lo splits (float4 in, 4x bf16 out)
// ---------------------------------------------------------------------------
__global__ void k_split4(const float4* __restrict__ src, v4bf* __restrict__ hi,
                         v4bf* __restrict__ lo, size_t n4) {
    size_t i = (size_t)blockIdx.x * blockDim.x + threadIdx.x;
    if (i >= n4) return;
    float4 v = src[i];
    float f[4] = {v.x, v.y, v.z, v.w};
    v4bf h, e;
#pragma unroll
    for (int j = 0; j < 4; ++j) {
        bf16 hb = (bf16)f[j];
        h[j] = hb;
        e[j] = (bf16)(f[j] - (float)hb);
    }
    hi[i] = h;
    lo[i] = e;
}

// h = relu(acc * scale), split into bf16 hi/lo for the next WMMA layer
__global__ void k_relu_mean_split4(const float4* __restrict__ accp, v4bf* __restrict__ hh,
                                   v4bf* __restrict__ he, size_t n4, float scale) {
    size_t i = (size_t)blockIdx.x * blockDim.x + threadIdx.x;
    if (i >= n4) return;
    float4 v = accp[i];
    float f[4] = {v.x, v.y, v.z, v.w};
    v4bf h, e;
#pragma unroll
    for (int j = 0; j < 4; ++j) {
        float vv = fmaxf(f[j] * scale, 0.0f);
        bf16 hb = (bf16)vv;
        h[j] = hb;
        e[j] = (bf16)(vv - (float)hb);
    }
    hh[i] = h;
    he[i] = e;
}

// W[r][K][M] fp32  ->  Wt[r][M][K] bf16 hi/lo (transposed for contiguous B loads)
__global__ void k_wsplit(const float* __restrict__ W, bf16* __restrict__ th,
                         bf16* __restrict__ te, int K, int M) {
    int idx = blockIdx.x * blockDim.x + threadIdx.x;
    if (idx >= K * M) return;
    int r = blockIdx.y;
    int k = idx / M, m = idx % M;
    float v = W[(size_t)r * K * M + idx];
    bf16 h = (bf16)v;
    th[(size_t)r * K * M + (size_t)m * K + k] = h;
    te[(size_t)r * K * M + (size_t)m * K + k] = (bf16)(v - (float)h);
}

// ---------------------------------------------------------------------------
// Vectorized accumulator utilities
// ---------------------------------------------------------------------------
__global__ void k_zero4(float4* __restrict__ p, size_t n4) {
    size_t i = (size_t)blockIdx.x * blockDim.x + threadIdx.x;
    if (i < n4) p[i] = make_float4(0.f, 0.f, 0.f, 0.f);
}

__global__ void k_scale4(float4* __restrict__ p, float s, size_t n4) {
    size_t i = (size_t)blockIdx.x * blockDim.x + threadIdx.x;
    if (i < n4) {
        float4 v = p[i];
        p[i] = make_float4(v.x * s, v.y * s, v.z * s, v.w * s);
    }
}

// ---------------------------------------------------------------------------
// Host launcher
// ---------------------------------------------------------------------------
extern "C" void kernel_launch(void* const* d_in, const int* in_sizes, int n_in,
                              void* d_out, int out_size, void* d_ws, size_t ws_size,
                              hipStream_t stream) {
    (void)in_sizes; (void)n_in; (void)out_size; (void)ws_size;
    const float* x  = (const float*)d_in[0];
    const int*   ei = (const int*)  d_in[1];   // [R][2][E]
    const float* W1 = (const float*)d_in[2];   // [R][DIN][DHID]
    const float* b1 = (const float*)d_in[3];   // [R][DHID]
    const float* W2 = (const float*)d_in[4];   // [R][DHID][DOUT]
    const float* b2 = (const float*)d_in[5];   // [R][DOUT]
    float* out = (float*)d_out;                // [N][DOUT]

    // --- workspace carve-up (256B aligned) ---
    char* ws = (char*)d_ws;
    size_t off = 0;
    auto carve = [&](size_t bytes) -> void* {
        void* p = ws + off;
        off += (bytes + 255) & ~(size_t)255;
        return p;
    };
    float* dinv = (float*)carve((size_t)NREL * NNODES * sizeof(float));
    bf16*  xh   = (bf16*) carve((size_t)NNODES * DIN * sizeof(bf16));   // reused as h_hi
    bf16*  xe   = (bf16*) carve((size_t)NNODES * DIN * sizeof(bf16));   // reused as h_lo
    bf16*  w1h  = (bf16*) carve((size_t)NREL * DIN * DHID * sizeof(bf16));
    bf16*  w1e  = (bf16*) carve((size_t)NREL * DIN * DHID * sizeof(bf16));
    bf16*  w2h  = (bf16*) carve((size_t)NREL * DHID * DOUT * sizeof(bf16));
    bf16*  w2e  = (bf16*) carve((size_t)NREL * DHID * DOUT * sizeof(bf16));
    float* xl   = (float*)carve((size_t)NNODES * DHID * sizeof(float)); // per-relation linear out
    float* acc  = (float*)carve((size_t)NNODES * DHID * sizeof(float)); // layer-1 accumulator

    const int T = 256;
    // 1) degrees -> dinv  (deg = in-degree(dst) + 1 self-loop; identical across layers)
    int ndeg = NREL * NNODES;
    k_fill1<<<(ndeg + T - 1) / T, T, 0, stream>>>(dinv, ndeg);
    dim3 gdc((NEDGES + T - 1) / T, NREL);
    k_deg_count<<<gdc, T, 0, stream>>>(ei, dinv, NEDGES, NNODES);
    k_rsqrt<<<(ndeg + T - 1) / T, T, 0, stream>>>(dinv, ndeg);

    // 2) bf16 hi/lo splits of x and transposed weights
    size_t nx4 = (size_t)NNODES * DIN / 4;
    k_split4<<<(unsigned)((nx4 + T - 1) / T), T, 0, stream>>>(
        (const float4*)x, (v4bf*)xh, (v4bf*)xe, nx4);
    dim3 gw1((DIN * DHID + T - 1) / T, NREL);
    k_wsplit<<<gw1, T, 0, stream>>>(W1, w1h, w1e, DIN, DHID);
    dim3 gw2((DHID * DOUT + T - 1) / T, NREL);
    k_wsplit<<<gw2, T, 0, stream>>>(W2, w2h, w2e, DHID, DOUT);

    // 3) layer 1: sum over relations of GCNConv (GEMM w/ fused self-loop+bias,
    //    then per-edge scatter), then relu(mean)
    size_t nacc4 = (size_t)NNODES * DHID / 4;
    k_zero4<<<(unsigned)((nacc4 + T - 1) / T), T, 0, stream>>>((float4*)acc, nacc4);
    for (int r = 0; r < NREL; ++r) {
        gcn_gemm_wmma_bf16<<<NNODES / 16, (DHID / 16) * 32, 0, stream>>>(
            xh, xe, w1h + (size_t)r * DIN * DHID, w1e + (size_t)r * DIN * DHID,
            dinv + (size_t)r * NNODES, b1 + (size_t)r * DHID,
            xl, acc, DIN, DHID);
        const int* src = ei + (size_t)r * 2 * NEDGES;
        const int* dst = src + NEDGES;
        k_edge_scatter128<<<(NEDGES + 7) / 8, T, 0, stream>>>(
            xl, dinv + (size_t)r * NNODES, src, dst, acc, NEDGES);
    }
    k_relu_mean_split4<<<(unsigned)((nacc4 + T - 1) / T), T, 0, stream>>>(
        (const float4*)acc, (v4bf*)xh, (v4bf*)xe, nacc4, 1.0f / NREL);

    // 4) layer 2 accumulates straight into d_out
    size_t nout4 = (size_t)NNODES * DOUT / 4;
    k_zero4<<<(unsigned)((nout4 + T - 1) / T), T, 0, stream>>>((float4*)out, nout4);
    for (int r = 0; r < NREL; ++r) {
        gcn_gemm_wmma_bf16<<<NNODES / 16, (DOUT / 16) * 32, 0, stream>>>(
            xh, xe, w2h + (size_t)r * DHID * DOUT, w2e + (size_t)r * DHID * DOUT,
            dinv + (size_t)r * NNODES, b2 + (size_t)r * DOUT,
            xl, out, DHID, DOUT);
        const int* src = ei + (size_t)r * 2 * NEDGES;
        const int* dst = src + NEDGES;
        k_edge_scatter64<<<(NEDGES + 7) / 8, T, 0, stream>>>(
            xl, dinv + (size_t)r * NNODES, src, dst, out, NEDGES);
    }
    k_scale4<<<(unsigned)((nout4 + T - 1) / T), T, 0, stream>>>((float4*)out, 1.0f / NREL, nout4);
}
